// SupConLoss_62878321213708
// MI455X (gfx1250) — compile-verified, matrix operand order checked by reading, not run on hardware
//
#include <hip/hip_runtime.h>
#include <hip/hip_bf16.h>

// ---------------------------------------------------------------------------
// SupCon loss, N=8192, D=128.
//   loss = -mean_i [ S_pos_i / cnt_i - log(rowexp_i + 1e-9) ]   (0 if cnt==0)
// Gram matrix via v_wmma_f32_16x16x32_bf16 with bf16 hi/lo error-corrected
// split (hi*hi + hi*lo + lo*hi) ~= fp32 precision, fp32 accumulate.
// ---------------------------------------------------------------------------

typedef __bf16 bf16x8  __attribute__((ext_vector_type(8)));
typedef __bf16 bf16x16 __attribute__((ext_vector_type(16)));
typedef float  v8f     __attribute__((ext_vector_type(8)));

#define MT      2     // 16-row tiles per wave (wave owns 32 rows)
#define KB      4     // K blocks of 32 (D == 128)
#define CHUNKS 16     // column-split factor (per-chunk partials, deterministic)
#define INV_T  10.0f  // 1 / temperature

// Load 16 bf16 for one lane of a 16x32 bf16 WMMA operand.
// Per ISA (16-bit A-matrix 16x32): lane half 0 -> K {koff..koff+7, 16+koff..},
// with koff = 8*(lane>=16). Elements 0..7 at p, 8..15 at p+16.
__device__ __forceinline__ bf16x16 ldv16(const __bf16* __restrict__ p) {
    bf16x8 a = *(const bf16x8*)p;
    bf16x8 b = *(const bf16x8*)(p + 16);
    return __builtin_shufflevector(a, b, 0,1,2,3,4,5,6,7,8,9,10,11,12,13,14,15);
}

__device__ __forceinline__ v8f wmma_bf16(bf16x16 a, bf16x16 b, v8f c) {
    return __builtin_amdgcn_wmma_f32_16x16x32_bf16(
        /*neg_a=*/false, a, /*neg_b=*/false, b,
        /*c_mod=*/(short)0, c, /*reuse_a=*/false, /*reuse_b=*/false);
}

// ---------------------------------------------------------------------------
// Kernel 1: L2-normalize each row, split into bf16 hi + bf16 lo residual.
// One block (D threads) per row.
// ---------------------------------------------------------------------------
__global__ void nrm_split_kernel(const float* __restrict__ f,
                                 __bf16* __restrict__ hi,
                                 __bf16* __restrict__ lo, int D) {
    const int row = blockIdx.x;
    const int t   = threadIdx.x;
    float x  = f[(size_t)row * D + t];
    float ss = x * x;
    #pragma unroll
    for (int m = 16; m >= 1; m >>= 1) ss += __shfl_xor(ss, m, 32);
    __shared__ float wsum[32];
    const int wid = t >> 5, lane = t & 31, nw = blockDim.x >> 5;
    if (lane == 0) wsum[wid] = ss;
    __syncthreads();
    float tot = 0.f;
    for (int w = 0; w < nw; ++w) tot += wsum[w];
    const float rn = 1.0f / sqrtf(tot);
    const float y  = x * rn;
    const __bf16 h = (__bf16)y;
    const float  r = y - (float)h;
    hi[(size_t)row * D + t] = h;
    lo[(size_t)row * D + t] = (__bf16)r;
}

// ---------------------------------------------------------------------------
// Kernel 2: Gram-matrix sweep. Each wave: 32 rows x one column chunk.
// Per 16x16 tile: 12 WMMAs (bfx3 over 4 K-blocks), then online exp/pos
// accumulation per lane, final in-wave shuffle reduction over the N dim.
// ---------------------------------------------------------------------------
__global__ void __launch_bounds__(256, 1)
gram_kernel(const __bf16* __restrict__ hi, const __bf16* __restrict__ lo,
            const int* __restrict__ labels,
            float* __restrict__ pExp, float* __restrict__ pPos,
            float* __restrict__ pCnt, int N, int D) {
    const int lane  = threadIdx.x & 31;
    const int gwave = blockIdx.x * (blockDim.x >> 5) + (threadIdx.x >> 5);

    const int rowTiles = N / (16 * MT);           // waves along rows
    const int rowWave  = gwave % rowTiles;
    const int chunk    = gwave / rowTiles;
    if (chunk >= CHUNKS) return;

    const int m0   = rowWave * 16 * MT;
    const int l16  = lane & 15;
    const int half = lane >> 4;
    const int koff = half * 8;

    // Resident A tiles (hi and lo residual), 16x32 layout per K-block.
    bf16x16 aH[MT][KB], aL[MT][KB];
    #pragma unroll
    for (int t = 0; t < MT; ++t) {
        const int arow = m0 + t * 16 + l16;
        const __bf16* ph = hi + (size_t)arow * D + koff;
        const __bf16* pl = lo + (size_t)arow * D + koff;
        #pragma unroll
        for (int kb = 0; kb < KB; ++kb) {
            aH[t][kb] = ldv16(ph + kb * 32);
            aL[t][kb] = ldv16(pl + kb * 32);
        }
    }

    // Row labels for the rows this lane's C fragments belong to
    // (C layout: VGPR r -> M = r + 8*half, N = l16).
    int labR[MT][8];
    #pragma unroll
    for (int t = 0; t < MT; ++t)
        #pragma unroll
        for (int r = 0; r < 8; ++r)
            labR[t][r] = labels[m0 + t * 16 + half * 8 + r];

    float aE[MT][8], aP[MT][8];
    int   aC[MT][8];
    #pragma unroll
    for (int t = 0; t < MT; ++t)
        #pragma unroll
        for (int r = 0; r < 8; ++r) { aE[t][r] = 0.f; aP[t][r] = 0.f; aC[t][r] = 0; }

    const int tilesPerChunk = (N / 16) / CHUNKS;
    const int c0base        = chunk * tilesPerChunk * 16;

    for (int ct = 0; ct < tilesPerChunk; ++ct) {
        const int col  = c0base + ct * 16 + l16;    // this lane's column
        const int labC = labels[col];
        const __bf16* bh = hi + (size_t)col * D + koff;
        const __bf16* bl = lo + (size_t)col * D + koff;

        if (ct + 1 < tilesPerChunk) {               // prefetch next tile's rows
            __builtin_prefetch(bh + 16 * D, 0, 1);
            __builtin_prefetch(bl + 16 * D, 0, 1);
        }

        v8f acc[MT];
        #pragma unroll
        for (int t = 0; t < MT; ++t) acc[t] = (v8f)0.f;

        #pragma unroll
        for (int kb = 0; kb < KB; ++kb) {
            const bf16x16 bH = ldv16(bh + kb * 32);
            const bf16x16 bL = ldv16(bl + kb * 32);
            #pragma unroll
            for (int t = 0; t < MT; ++t) {
                acc[t] = wmma_bf16(aH[t][kb], bH, acc[t]);  // hi*hi
                acc[t] = wmma_bf16(aH[t][kb], bL, acc[t]);  // hi*lo
                acc[t] = wmma_bf16(aL[t][kb], bH, acc[t]);  // lo*hi
            }
        }

        #pragma unroll
        for (int t = 0; t < MT; ++t) {
            #pragma unroll
            for (int r = 0; r < 8; ++r) {
                const float s   = acc[t][r] * INV_T;
                const int   row = m0 + t * 16 + half * 8 + r;
                const bool  off = (row != col);              // exclude diagonal
                const float e   = __expf(s);
                aE[t][r] += off ? e : 0.f;
                const bool m = off & (labR[t][r] == labC);
                aP[t][r] += m ? s : 0.f;
                aC[t][r] += m ? 1 : 0;
            }
        }
    }

    // Reduce across the 16 lanes sharing each row (masks 1..8 stay in-half).
    #pragma unroll
    for (int m = 8; m >= 1; m >>= 1) {
        #pragma unroll
        for (int t = 0; t < MT; ++t)
            #pragma unroll
            for (int r = 0; r < 8; ++r) {
                aE[t][r] += __shfl_xor(aE[t][r], m, 32);
                aP[t][r] += __shfl_xor(aP[t][r], m, 32);
                aC[t][r] += __shfl_xor(aC[t][r], m, 32);
            }
    }

    if (l16 == 0) {
        #pragma unroll
        for (int t = 0; t < MT; ++t)
            #pragma unroll
            for (int r = 0; r < 8; ++r) {
                const int row = m0 + t * 16 + half * 8 + r;
                const size_t o = (size_t)chunk * N + row;
                pExp[o] = aE[t][r];
                pPos[o] = aP[t][r];
                pCnt[o] = (float)aC[t][r];
            }
    }
}

// ---------------------------------------------------------------------------
// Kernel 3: combine chunk partials, compute per-row term, reduce to loss.
// Single block (deterministic).
// ---------------------------------------------------------------------------
__global__ void finalize_kernel(const float* __restrict__ pExp,
                                const float* __restrict__ pPos,
                                const float* __restrict__ pCnt,
                                float* __restrict__ out, int N) {
    float s = 0.f;
    for (int i = threadIdx.x; i < N; i += blockDim.x) {
        float E = 0.f, P = 0.f, C = 0.f;
        #pragma unroll
        for (int c = 0; c < CHUNKS; ++c) {
            E += pExp[(size_t)c * N + i];
            P += pPos[(size_t)c * N + i];
            C += pCnt[(size_t)c * N + i];
        }
        const float L = logf(E + 1e-9f);
        s += (C > 0.f) ? (P - C * L) / C : 0.f;
    }
    #pragma unroll
    for (int m = 16; m >= 1; m >>= 1) s += __shfl_xor(s, m, 32);
    __shared__ float red[8];
    const int wid = threadIdx.x >> 5, lane = threadIdx.x & 31;
    if (lane == 0) red[wid] = s;
    __syncthreads();
    if (threadIdx.x == 0) {
        float tot = 0.f;
        const int nw = blockDim.x >> 5;
        for (int w = 0; w < nw; ++w) tot += red[w];
        out[0] = -tot / (float)N;
    }
}

// ---------------------------------------------------------------------------
extern "C" void kernel_launch(void* const* d_in, const int* in_sizes, int n_in,
                              void* d_out, int out_size, void* d_ws, size_t ws_size,
                              hipStream_t stream) {
    const float* feats  = (const float*)d_in[0];
    const int*   labels = (const int*)d_in[1];
    const int N = in_sizes[1];             // 8192
    const int D = in_sizes[0] / N;         // 128

    char* ws = (char*)d_ws;
    __bf16* hi = (__bf16*)ws;
    __bf16* lo = (__bf16*)(ws + (size_t)N * D * sizeof(__bf16));
    float* pExp = (float*)(ws + (size_t)N * D * 2 * sizeof(__bf16));
    float* pPos = pExp + (size_t)CHUNKS * N;
    float* pCnt = pPos + (size_t)CHUNKS * N;

    // 1) normalize + bf16 hi/lo split
    nrm_split_kernel<<<N, D, 0, stream>>>(feats, hi, lo, D);

    // 2) WMMA Gram sweep: (N / (16*MT)) row-waves x CHUNKS column chunks
    const int rowTiles = N / (16 * MT);
    const int totWaves = rowTiles * CHUNKS;
    const int wavesPerBlock = 8;           // 256 threads
    gram_kernel<<<totWaves / wavesPerBlock, wavesPerBlock * 32, 0, stream>>>(
        hi, lo, labels, pExp, pPos, pCnt, N, D);

    // 3) finalize
    finalize_kernel<<<1, 256, 0, stream>>>(pExp, pPos, pCnt, (float*)d_out, N);
}
